// CrimeHeteroGNN_80135499809347
// MI455X (gfx1250) — matrix operand order repaired and checked by reading, not compiled, not run on hardware
//
#include <hip/hip_runtime.h>
#include <cstddef>

// ---------------------------------------------------------------------------
// CDNA5 (gfx1250) hetero-GraphSAGE inference.
// wave32; fused WMMA f32_16x16x32_bf16 SAGE GEMM (mean@Wl^T + b + x@Wr^T in
// one pass); hardware f32 global atomics for edge scatter (all node tensors
// fit in the 192MB L2).
// ---------------------------------------------------------------------------

typedef __attribute__((ext_vector_type(16))) __bf16 v16bf;
typedef __attribute__((ext_vector_type(8)))  float  v8f;

#define ATOMIC_FADD(p, v) __hip_atomic_fetch_add((p), (v), __ATOMIC_RELAXED, __HIP_MEMORY_SCOPE_AGENT)

// ---------------- utility kernels ----------------

__global__ void zero_kernel(float* __restrict__ p, size_t n) {
  size_t i = (size_t)blockIdx.x * blockDim.x + threadIdx.x;
  size_t stride = (size_t)gridDim.x * blockDim.x;
  for (; i < n; i += stride) p[i] = 0.0f;
}

// out[node][0:64] = emb[idx[node]][0:64]; thread = (node, 16B chunk)
__global__ void gather_kernel(const int* __restrict__ idx, const float* __restrict__ emb,
                              float* __restrict__ out, int n) {
  int t = blockIdx.x * blockDim.x + threadIdx.x;
  if (t >= n * 16) return;
  int node = t >> 4;
  int c = (t & 15) * 4;
  int e = idx[node];
  *(float4*)(out + (size_t)node * 64 + c) = *(const float4*)(emb + (size_t)e * 64 + c);
}

__global__ void count_kernel(const int* __restrict__ dst, int* __restrict__ cnt, int E) {
  int t = blockIdx.x * blockDim.x + threadIdx.x;
  if (t < E) atomicAdd(&cnt[dst[t]], 1);
}

// one wave per edge: 32 lanes x float2 = one 64-float feature row
__global__ void scatter_kernel(const int* __restrict__ src, const int* __restrict__ dst,
                               const float* __restrict__ feat, float* __restrict__ agg, int E) {
  int gid = blockIdx.x * blockDim.x + threadIdx.x;
  int e = gid >> 5;
  int lane = gid & 31;
  if (e >= E) return;
  int s = src[e];
  int d = dst[e];
  const float2 v = *(const float2*)(feat + (size_t)s * 64 + lane * 2);
  float* base = agg + (size_t)d * 64 + lane * 2;
  ATOMIC_FADD(base + 0, v.x);
  ATOMIC_FADD(base + 1, v.y);
}

// feat = relu(acc * invdiv) + feat
__global__ void update_kernel(float* __restrict__ feat, const float* __restrict__ acc,
                              size_t n, float invdiv) {
  size_t i = (size_t)blockIdx.x * blockDim.x + threadIdx.x;
  size_t stride = (size_t)gridDim.x * blockDim.x;
  for (; i < n; i += stride) feat[i] = fmaxf(acc[i] * invdiv, 0.0f) + feat[i];
}

// ---------------- fused SAGE WMMA GEMM ----------------
// out[n,64] += (Agg .* 1/cnt) @ Wl^T + bias + X @ Wr^T
// 128 threads = 4 waves; each wave owns a 16-row tile; block covers 64 rows.
// LDS: Wl (16KB) + Wr (16KB) + 4 x per-wave 16x64 A slot (16KB) = 48KB.
#define GW 4

__device__ __forceinline__ void build_afrag(const float* __restrict__ ar, float scale,
                                            int koff, v16bf& f0, v16bf& f1) {
  // 16-bit A 16x32 VGPR layout (ISA 7.12.2); f0 = K[0:32), f1 = K[32:64)
#pragma unroll
  for (int v = 0; v < 8; ++v) {
    int k = ((v & 4) ? 16 : 0) + koff + 2 * (v & 3);
    f0[2 * v]     = (__bf16)(ar[k] * scale);
    f0[2 * v + 1] = (__bf16)(ar[k + 1] * scale);
    f1[2 * v]     = (__bf16)(ar[32 + k] * scale);
    f1[2 * v + 1] = (__bf16)(ar[32 + k + 1] * scale);
  }
}

__device__ __forceinline__ void build_bfrag(const float* __restrict__ wr, int kb,
                                            v16bf& f0, v16bf& f1) {
  // B 32x16 16-bit layout: B[k][col] = W[col][k]; f0 = K[0:32), f1 = K[32:64)
#pragma unroll
  for (int v = 0; v < 8; ++v) {
    int k = ((v & 4) ? 16 : 0) + kb + 2 * (v & 3);
    f0[2 * v]     = (__bf16)wr[k];
    f0[2 * v + 1] = (__bf16)wr[k + 1];
    f1[2 * v]     = (__bf16)wr[32 + k];
    f1[2 * v + 1] = (__bf16)wr[32 + k + 1];
  }
}

__global__ __launch_bounds__(128) void sage_gemm_kernel(
    const float* __restrict__ Agg, const int* __restrict__ cnt,  // cnt==nullptr -> no mean scaling
    const float* __restrict__ X,                                 // root input
    const float* __restrict__ Wl, const float* __restrict__ Wr,  // [64][64] row-major (j = out col)
    const float* __restrict__ bias,
    float* __restrict__ out, int n) {
  __shared__ float ldsWl[64 * 64];
  __shared__ float ldsWr[64 * 64];
  __shared__ float ldsA[GW][16 * 64];

  const int tid = threadIdx.x;
  const int wave = tid >> 5;
  const int lane = tid & 31;

  for (int i = tid; i < 64 * 64; i += 128) {
    ldsWl[i] = Wl[i];
    ldsWr[i] = Wr[i];
  }

  const int r0 = (blockIdx.x * GW + wave) * 16;
  // stage this wave's 16x64 Agg tile (zero-padded past n)
  for (int i = lane; i < 256; i += 32) {
    int row = r0 + (i >> 4);
    float4 v = make_float4(0.f, 0.f, 0.f, 0.f);
    if (row < n) v = *(const float4*)(Agg + (size_t)row * 64 + (i & 15) * 4);
    *(float4*)&ldsA[wave][i * 4] = v;
  }
  __syncthreads();                 // ldsW visibility across waves
  if (r0 >= n) return;             // wave-uniform; EXEC all-1s for surviving waves

  const int m = lane & 15;
  const int koff = (lane & 16) ? 8 : 0;
  const float* ar = &ldsA[wave][m * 64];

  float scale = 1.0f;
  if (cnt != nullptr) {
    int rowm = r0 + m;
    int c = (rowm < n) ? cnt[rowm] : 1;
    if (c < 1) c = 1;
    scale = 1.0f / (float)c;
  }
  v16bf a0, a1;
  build_afrag(ar, scale, koff, a0, a1);

  // restage X tile into the same per-wave slot (LDS ops are in-order per wave)
  for (int i = lane; i < 256; i += 32) {
    int row = r0 + (i >> 4);
    float4 v = make_float4(0.f, 0.f, 0.f, 0.f);
    if (row < n) v = *(const float4*)(X + (size_t)row * 64 + (i & 15) * 4);
    *(float4*)&ldsA[wave][i * 4] = v;
  }
  v16bf a2, a3;
  build_afrag(ar, 1.0f, koff, a2, a3);

  const int nc = lane & 15;
  const int kb = (lane & 16) ? 8 : 0;
  const int mb = (lane & 16) ? 8 : 0;
  const bool full = (r0 + 16 <= n);   // wave-uniform fast path

#pragma unroll
  for (int ntile = 0; ntile < 4; ++ntile) {
    const int col = ntile * 16 + nc;
    v16bf bl0, bl1, br0, br1;
    build_bfrag(&ldsWl[col * 64], kb, bl0, bl1);
    build_bfrag(&ldsWr[col * 64], kb, br0, br1);

    const float bv = (bias != nullptr) ? bias[col] : 0.0f;
    float* cbase = out + (size_t)(r0 + mb) * 64 + col;
    v8f c;
    if (full) {
#pragma unroll
      for (int v = 0; v < 8; ++v) c[v] = cbase[(size_t)v * 64] + bv;
    } else {
#pragma unroll
      for (int v = 0; v < 8; ++v) c[v] = (r0 + mb + v < n) ? (cbase[(size_t)v * 64] + bv) : 0.0f;
    }

    c = __builtin_amdgcn_wmma_f32_16x16x32_bf16(false, a0, false, bl0, (short)0, c, false, false);
    c = __builtin_amdgcn_wmma_f32_16x16x32_bf16(false, a1, false, bl1, (short)0, c, false, false);
    c = __builtin_amdgcn_wmma_f32_16x16x32_bf16(false, a2, false, br0, (short)0, c, false, false);
    c = __builtin_amdgcn_wmma_f32_16x16x32_bf16(false, a3, false, br1, (short)0, c, false, false);

    if (full) {
#pragma unroll
      for (int v = 0; v < 8; ++v) cbase[(size_t)v * 64] = c[v];
    } else {
#pragma unroll
      for (int v = 0; v < 8; ++v) if (r0 + mb + v < n) cbase[(size_t)v * 64] = c[v];
    }
  }
}

// ---------------- readout ----------------

// column sums of a[n][64] into gout[0:64] (atomic accumulate)
__global__ void colsum_kernel(const float* __restrict__ a, float* __restrict__ gout, int n) {
  int t = blockIdx.x * blockDim.x + threadIdx.x;
  int col = t & 63;
  int row = t >> 6;
  int rstride = (gridDim.x * blockDim.x) >> 6;
  float s = 0.0f;
  for (; row < n; row += rstride) s += a[(size_t)row * 64 + col];
  ATOMIC_FADD(&gout[col], s);
}

// single block, 192 threads: crime logits head
__global__ void crime_head_kernel(const float* __restrict__ gsum,
                                  const float* __restrict__ Wc1, const float* __restrict__ bc1,
                                  const float* __restrict__ Wc2, const float* __restrict__ bc2,
                                  float n_p, float n_o, float n_l,
                                  float* __restrict__ out) {
  __shared__ float g[192];
  __shared__ float h[64];
  int t = threadIdx.x;
  if (t < 192) {
    float inv = (t < 64) ? (1.0f / n_p) : ((t < 128) ? (1.0f / n_o) : (1.0f / n_l));
    g[t] = gsum[t] * inv;
  }
  __syncthreads();
  if (t < 64) {
    float acc = bc1[t];
    for (int k = 0; k < 192; ++k) acc += g[k] * Wc1[t * 192 + k];
    h[t] = fmaxf(acc, 0.0f);
  }
  __syncthreads();
  if (t < 20) {
    float acc = bc2[t];
    for (int j = 0; j < 64; ++j) acc += h[j] * Wc2[t * 64 + j];
    out[t] = acc;
  }
}

// per-person 64 -> 32 -> 1 MLP
__global__ __launch_bounds__(256) void suspect_kernel(
    const float* __restrict__ p,
    const float* __restrict__ Ws1, const float* __restrict__ bs1,
    const float* __restrict__ Ws2, const float* __restrict__ bs2,
    float* __restrict__ out, int n) {
  __shared__ float w1[32 * 64];
  __shared__ float b1[32];
  __shared__ float w2[32];
  int tid = threadIdx.x;
  for (int i = tid; i < 32 * 64; i += 256) w1[i] = Ws1[i];
  if (tid < 32) { b1[tid] = bs1[tid]; w2[tid] = Ws2[tid]; }
  __syncthreads();
  int i = blockIdx.x * 256 + tid;
  if (i >= n) return;
  float x[64];
  const float4* pr = (const float4*)(p + (size_t)i * 64);
#pragma unroll
  for (int q = 0; q < 16; ++q) {
    float4 v = pr[q];
    x[4 * q] = v.x; x[4 * q + 1] = v.y; x[4 * q + 2] = v.z; x[4 * q + 3] = v.w;
  }
  float score = bs2[0];
#pragma unroll 4
  for (int j = 0; j < 32; ++j) {
    float a = b1[j];
#pragma unroll
    for (int k = 0; k < 64; ++k) a += x[k] * w1[j * 64 + k];
    score += fmaxf(a, 0.0f) * w2[j];
  }
  out[i] = score;
}

// ---------------- host orchestration ----------------

static inline int cdiv(long long a, long long b) { return (int)((a + b - 1) / b); }

extern "C" void kernel_launch(void* const* d_in, const int* in_sizes, int n_in,
                              void* d_out, int out_size, void* d_ws, size_t ws_size,
                              hipStream_t stream) {
  const int H = 64;
  const int N_P = in_sizes[0], N_O = in_sizes[1], N_L = in_sizes[2];
  const int E = in_sizes[3];

  const int*   x_person = (const int*)d_in[0];
  const int*   x_object = (const int*)d_in[1];
  const int*   x_locat  = (const int*)d_in[2];
  const int*   acts_src = (const int*)d_in[3];
  const int*   acts_dst = (const int*)d_in[4];
  const int*   uses_src = (const int*)d_in[5];
  const int*   uses_dst = (const int*)d_in[6];
  const int*   at_src   = (const int*)d_in[7];
  const int*   at_dst   = (const int*)d_in[8];
  const float* person_emb = (const float*)d_in[9];
  const float* object_emb = (const float*)d_in[10];
  const float* locat_emb  = (const float*)d_in[11];
  const float* Wl  = (const float*)d_in[12];   // [3][5][64][64]
  const float* bl  = (const float*)d_in[13];   // [3][5][64]
  const float* Wr  = (const float*)d_in[14];   // [3][5][64][64]
  const float* Wc1 = (const float*)d_in[15];
  const float* bc1 = (const float*)d_in[16];
  const float* Wc2 = (const float*)d_in[17];
  const float* bc2 = (const float*)d_in[18];
  const float* Ws1 = (const float*)d_in[19];
  const float* bs1 = (const float*)d_in[20];
  const float* Ws2 = (const float*)d_in[21];
  const float* bs2 = (const float*)d_in[22];
  float* out = (float*)d_out;
  (void)ws_size; (void)n_in; (void)out_size;

  // workspace layout
  float* ws = (float*)d_ws;
  size_t cur = 0;
  float* p     = ws + cur; cur += (size_t)N_P * H;
  float* p_acc = ws + cur; cur += (size_t)N_P * H;
  float* o     = ws + cur; cur += (size_t)N_O * H;
  float* o_acc = ws + cur; cur += (size_t)N_O * H;
  float* l     = ws + cur; cur += (size_t)N_L * H;
  float* l_acc = ws + cur; cur += (size_t)N_L * H;
  float* agg   = ws + cur; cur += (size_t)N_P * H;   // reused, sized for largest dst set
  int* cnt_p0 = (int*)(ws + cur); cur += N_P;
  int* cnt_p2 = (int*)(ws + cur); cur += N_P;
  int* cnt_p4 = (int*)(ws + cur); cur += N_P;
  int* cnt_o  = (int*)(ws + cur); cur += N_O;
  int* cnt_l  = (int*)(ws + cur); cur += N_L;
  float* gsum = ws + cur; cur += 192;

  const int ZB = 1024;                 // grid-stride zero blocks
  const int B = 256;

  // 1. embedding gathers
  gather_kernel<<<cdiv((long long)N_P * 16, B), B, 0, stream>>>(x_person, person_emb, p, N_P);
  gather_kernel<<<cdiv((long long)N_O * 16, B), B, 0, stream>>>(x_object, object_emb, o, N_O);
  gather_kernel<<<cdiv((long long)N_L * 16, B), B, 0, stream>>>(x_locat,  locat_emb,  l, N_L);

  // 2. degree counts (edge-structure only; computed once per call)
  zero_kernel<<<ZB, B, 0, stream>>>((float*)cnt_p0, (size_t)(3 * N_P + N_O + N_L)); // contiguous
  count_kernel<<<cdiv(E, B), B, 0, stream>>>(acts_dst, cnt_p0, E);
  count_kernel<<<cdiv(E, B), B, 0, stream>>>(uses_src, cnt_p2, E);
  count_kernel<<<cdiv(E, B), B, 0, stream>>>(at_src,   cnt_p4, E);
  count_kernel<<<cdiv(E, B), B, 0, stream>>>(uses_dst, cnt_o,  E);
  count_kernel<<<cdiv(E, B), B, 0, stream>>>(at_dst,   cnt_l,  E);

  const int scat_blocks = cdiv((long long)E * 32, B);

  // 3. three SAGE layers
  for (int t = 0; t < 3; ++t) {
    const float* Wl_t = Wl + (size_t)t * 5 * 64 * 64;
    const float* bl_t = bl + (size_t)t * 5 * 64;
    const float* Wr_t = Wr + (size_t)t * 5 * 64 * 64;

    zero_kernel<<<ZB, B, 0, stream>>>(p_acc, (size_t)N_P * H);
    zero_kernel<<<ZB, B, 0, stream>>>(o_acc, (size_t)N_O * H);
    zero_kernel<<<ZB, B, 0, stream>>>(l_acc, (size_t)N_L * H);

    // rel 0: person <- person (acts_on); fused mean + bias + root
    zero_kernel<<<ZB, B, 0, stream>>>(agg, (size_t)N_P * H);
    scatter_kernel<<<scat_blocks, B, 0, stream>>>(acts_src, acts_dst, p, agg, E);
    sage_gemm_kernel<<<cdiv(N_P, 64), 128, 0, stream>>>(
        agg, cnt_p0, p, Wl_t + 0 * 4096, Wr_t + 0 * 4096, bl_t + 0 * 64, p_acc, N_P);

    // rel 2: person <- object (rev_uses)
    zero_kernel<<<ZB, B, 0, stream>>>(agg, (size_t)N_P * H);
    scatter_kernel<<<scat_blocks, B, 0, stream>>>(uses_dst, uses_src, o, agg, E);
    sage_gemm_kernel<<<cdiv(N_P, 64), 128, 0, stream>>>(
        agg, cnt_p2, p, Wl_t + 2 * 4096, Wr_t + 2 * 4096, bl_t + 2 * 64, p_acc, N_P);

    // rel 4: person <- location (rev_at)
    zero_kernel<<<ZB, B, 0, stream>>>(agg, (size_t)N_P * H);
    scatter_kernel<<<scat_blocks, B, 0, stream>>>(at_dst, at_src, l, agg, E);
    sage_gemm_kernel<<<cdiv(N_P, 64), 128, 0, stream>>>(
        agg, cnt_p4, p, Wl_t + 4 * 4096, Wr_t + 4 * 4096, bl_t + 4 * 64, p_acc, N_P);

    // rel 1: object <- person (uses)
    zero_kernel<<<ZB, B, 0, stream>>>(agg, (size_t)N_O * H);
    scatter_kernel<<<scat_blocks, B, 0, stream>>>(uses_src, uses_dst, p, agg, E);
    sage_gemm_kernel<<<cdiv(N_O, 64), 128, 0, stream>>>(
        agg, cnt_o, o, Wl_t + 1 * 4096, Wr_t + 1 * 4096, bl_t + 1 * 64, o_acc, N_O);

    // rel 3: location <- person (at)
    zero_kernel<<<ZB, B, 0, stream>>>(agg, (size_t)N_L * H);
    scatter_kernel<<<scat_blocks, B, 0, stream>>>(at_src, at_dst, p, agg, E);
    sage_gemm_kernel<<<cdiv(N_L, 64), 128, 0, stream>>>(
        agg, cnt_l, l, Wl_t + 3 * 4096, Wr_t + 3 * 4096, bl_t + 3 * 64, l_acc, N_L);

    // fused relu + residual updates (HeteroConv mean over 3 relations for person)
    update_kernel<<<ZB, B, 0, stream>>>(p, p_acc, (size_t)N_P * H, 1.0f / 3.0f);
    update_kernel<<<ZB, B, 0, stream>>>(o, o_acc, (size_t)N_O * H, 1.0f);
    update_kernel<<<ZB, B, 0, stream>>>(l, l_acc, (size_t)N_L * H, 1.0f);
  }

  // 4. global mean pooling
  zero_kernel<<<1, 256, 0, stream>>>(gsum, 192);
  colsum_kernel<<<512, B, 0, stream>>>(p, gsum + 0,   N_P);
  colsum_kernel<<<512, B, 0, stream>>>(o, gsum + 64,  N_O);
  colsum_kernel<<<512, B, 0, stream>>>(l, gsum + 128, N_L);

  // 5. crime head -> out[0:20]
  crime_head_kernel<<<1, 192, 0, stream>>>(gsum, Wc1, bc1, Wc2, bc2,
                                           (float)N_P, (float)N_O, (float)N_L, out);

  // 6. suspect head -> out[20 : 20+N_P]
  suspect_kernel<<<cdiv(N_P, 256), 256, 0, stream>>>(p, Ws1, bs1, Ws2, bs2, out + 20, N_P);
}
